// DDLTransformerBlock_28724741276188
// MI455X (gfx1250) — compile-verified, matrix operand order checked by reading, not run on hardware
//
#include <hip/hip_runtime.h>
#include <cstdint>

#define B_  2
#define T_  2048
#define D_  1024
#define H_  16
#define HD_ 64
#define FH_ 2752
#define M_  (B_*T_)

typedef __bf16 bf16;
typedef bf16  v16bf __attribute__((ext_vector_type(16)));
typedef bf16  v8bf  __attribute__((ext_vector_type(8)));
typedef float v8f   __attribute__((ext_vector_type(8)));

#if __has_builtin(__builtin_amdgcn_tensor_load_to_lds) && \
    __has_builtin(__builtin_amdgcn_s_wait_tensorcnt)
#define USE_TDM 1
typedef unsigned int v4u_ __attribute__((ext_vector_type(4)));
typedef int          v8i_ __attribute__((ext_vector_type(8)));
typedef int          v4i_ __attribute__((ext_vector_type(4)));
#endif

// ---------------------------------------------------------------- helpers

__device__ __forceinline__ v8f vzero8() {
  v8f z;
#pragma unroll
  for (int i = 0; i < 8; ++i) z[i] = 0.0f;
  return z;
}

// Load a 16x32 bf16 WMMA A/B fragment from a row-major (padded) tile.
__device__ __forceinline__ v16bf load_frag(const bf16* base, int stride) {
  const int lane = threadIdx.x & 31;
  const bf16* p = base + (lane & 15) * stride + ((lane >> 4) << 3);
  v16bf f;
  ((v8bf*)&f)[0] = *((const v8bf*)(p));
  ((v8bf*)&f)[1] = *((const v8bf*)(p + 16));
  return f;
}

__device__ __forceinline__ v8f wmma_bf16(v16bf a, v16bf b, v8f c) {
  return __builtin_amdgcn_wmma_f32_16x16x32_bf16(false, a, false, b, (short)0, c,
                                                 false, false);
}

__device__ __forceinline__ float sigmoidf_(float x) {
  return 1.0f / (1.0f + __expf(-x));
}

#ifdef USE_TDM
// Issue one TDM 2D tile load: rows x 32 bf16 elements, row stride = `stride`
// elements in global, LDS gets 64B rows padded by 16B (40-half stride).
// dim1 < tile1 rows beyond the tensor edge are zero-filled by hardware.
__device__ __forceinline__ void tdm_load_2d(const bf16* g, unsigned ldsOff,
                                            int dim1, int stride, int tile1) {
  const unsigned long long ga = (unsigned long long)(uintptr_t)g;
  v4u_ g0;
  g0[0] = 1u;                                    // count=1 (valid), user mode
  g0[1] = ldsOff;                                // lds_addr (bytes)
  g0[2] = (unsigned)ga;                          // global_addr[31:0]
  g0[3] = (unsigned)((ga >> 32) & 0x01FFFFFFull) // global_addr[56:32]
          | 0x80000000u;                         // type=2 ("image")
  v8i_ g1;
  g1[0] = (1 << 16)      // data_size = 2 bytes
        | (1 << 20)      // pad_enable
        | (3 << 22)      // pad_interval: 16 DWORDs (one 64B row)
        | (3 << 25);     // pad_amount: 4 DWORDs (16B) -> 80B LDS row stride
  g1[1] = (32 & 0xFFFF) << 16;                       // tensor_dim0 = 32
  g1[2] = (int)(((unsigned)dim1 & 0xFFFFu) << 16);   // tensor_dim1 lo
  g1[3] = (int)((((unsigned)dim1 >> 16) & 0xFFFFu)   // tensor_dim1 hi
                | (32u << 16));                      // tile_dim0 = 32
  g1[4] = tile1 & 0xFFFF;                            // tile_dim1 (tile_dim2=0)
  g1[5] = stride;                                    // tensor_dim0_stride lo32
  g1[6] = 0;                                         // stride hi / dim1_stride
  g1[7] = 0;
  v4i_ gz;
  gz[0] = 0; gz[1] = 0; gz[2] = 0; gz[3] = 0;
#if __clang_major__ >= 23
  v8i_ gz8;
#pragma unroll
  for (int i = 0; i < 8; ++i) gz8[i] = 0;
  __builtin_amdgcn_tensor_load_to_lds(g0, g1, gz, gz, gz8, 0);
#else
  __builtin_amdgcn_tensor_load_to_lds(g0, g1, gz, gz, 0);
#endif
}
#endif  // USE_TDM

// ---------------------------------------------------------------- converts

__global__ void convert_f32_bf16(const float* __restrict__ src,
                                 bf16* __restrict__ dst, int n) {
  int i = blockIdx.x * blockDim.x + threadIdx.x;
  int st = gridDim.x * blockDim.x;
  for (; i < n; i += st) dst[i] = (bf16)src[i];
}

// ---------------------------------------------------------------- rmsnorm -> bf16

__global__ __launch_bounds__(256) void rmsnorm_bf16_kernel(
    const float* __restrict__ x, const float* __restrict__ w,
    bf16* __restrict__ out) {
  const int row = blockIdx.x;
  const float* xr = x + (size_t)row * D_;
  float ss = 0.0f;
  for (int i = threadIdx.x; i < D_; i += 256) { float v = xr[i]; ss += v * v; }
#pragma unroll
  for (int off = 16; off; off >>= 1) ss += __shfl_xor(ss, off);
  __shared__ float red[8];
  if ((threadIdx.x & 31) == 0) red[threadIdx.x >> 5] = ss;
  __syncthreads();
  float tot = 0.0f;
#pragma unroll
  for (int wdx = 0; wdx < 8; ++wdx) tot += red[wdx];
  const float r = rsqrtf(tot * (1.0f / D_) + 1e-6f);
  for (int i = threadIdx.x; i < D_; i += 256)
    out[(size_t)row * D_ + i] = (bf16)(xr[i] * r * w[i]);
}

// ---------------------------------------------------------------- WMMA GEMM
// C[M,N] = A[M,K] (bf16 row-major) x Bw[N,K]^T (bf16 row-major weights)
// WG: 256 threads (8 waves). Tile 128x256x32; wave tile 64x64 (4x4 accums).
// TDM path: double-buffered LDS, next k-tile DMA'd while WMMAs run.

template <typename OutT>
__global__ __launch_bounds__(256) void gemm_bf16_kernel(
    const bf16* __restrict__ A, const bf16* __restrict__ Bw,
    OutT* __restrict__ C, int N, int K) {
  __shared__ bf16 As[2][128 * 40];
  __shared__ bf16 Bs[2][256 * 40];
  const int m0 = blockIdx.y * 128;
  const int n0 = blockIdx.x * 256;
  const int t = threadIdx.x;
  const int lane = t & 31, wid = t >> 5;
  const int wm = wid >> 2, wn = wid & 3;
  const int half = lane >> 4, col0 = lane & 15;

  v8f acc[4][4];
#pragma unroll
  for (int mi = 0; mi < 4; ++mi)
#pragma unroll
    for (int ni = 0; ni < 4; ++ni) acc[mi][ni] = vzero8();

#ifdef USE_TDM
  const int rowsB = N - n0;  // TDM zero-fills rows beyond tensor_dim1
  if (wid == 0) {
    tdm_load_2d(A + (size_t)m0 * K, (unsigned)(uintptr_t)&As[0][0], 128, K, 128);
    tdm_load_2d(Bw + (size_t)n0 * K, (unsigned)(uintptr_t)&Bs[0][0], rowsB, K, 256);
  }
  int buf = 0;
  for (int k0 = 0; k0 < K; k0 += 32) {
    if (wid == 0) {
      if (k0 + 32 < K) {
        tdm_load_2d(A + (size_t)m0 * K + (k0 + 32),
                    (unsigned)(uintptr_t)&As[buf ^ 1][0], 128, K, 128);
        tdm_load_2d(Bw + (size_t)n0 * K + (k0 + 32),
                    (unsigned)(uintptr_t)&Bs[buf ^ 1][0], rowsB, K, 256);
        __builtin_amdgcn_s_wait_tensorcnt((short)2);  // current tile done
      } else {
        __builtin_amdgcn_s_wait_tensorcnt((short)0);
      }
    }
    __syncthreads();  // current tile visible to all waves

    v16bf af[4], bfr[4];
#pragma unroll
    for (int mi = 0; mi < 4; ++mi)
      af[mi] = load_frag(&As[buf][(wm * 64 + mi * 16) * 40], 40);
#pragma unroll
    for (int ni = 0; ni < 4; ++ni)
      bfr[ni] = load_frag(&Bs[buf][(wn * 64 + ni * 16) * 40], 40);
#pragma unroll
    for (int mi = 0; mi < 4; ++mi)
#pragma unroll
      for (int ni = 0; ni < 4; ++ni)
        acc[mi][ni] = wmma_bf16(af[mi], bfr[ni], acc[mi][ni]);

    __syncthreads();  // all reads done before this buffer is DMA'd again
    buf ^= 1;
  }
#else
  for (int k0 = 0; k0 < K; k0 += 32) {
#pragma unroll
    for (int i = 0; i < 2; ++i) {  // A: 512 chunks of 8 halves
      const int c = t + i * 256;
      const int r = c >> 2, cid = c & 3;
      *(v8bf*)&As[0][r * 40 + cid * 8] =
          *(const v8bf*)(A + (size_t)(m0 + r) * K + k0 + cid * 8);
    }
#pragma unroll
    for (int i = 0; i < 4; ++i) {  // B: 1024 chunks of 8 halves
      const int c = t + i * 256;
      const int r = c >> 2, cid = c & 3;
      const int n = n0 + r;
      v8bf vb;
      if (n < N) {
        vb = *(const v8bf*)(Bw + (size_t)n * K + k0 + cid * 8);
      } else {
#pragma unroll
        for (int j = 0; j < 8; ++j) vb[j] = (bf16)0.0f;
      }
      *(v8bf*)&Bs[0][r * 40 + cid * 8] = vb;
    }
    __syncthreads();
    v16bf af[4], bfr[4];
#pragma unroll
    for (int mi = 0; mi < 4; ++mi)
      af[mi] = load_frag(&As[0][(wm * 64 + mi * 16) * 40], 40);
#pragma unroll
    for (int ni = 0; ni < 4; ++ni)
      bfr[ni] = load_frag(&Bs[0][(wn * 64 + ni * 16) * 40], 40);
#pragma unroll
    for (int mi = 0; mi < 4; ++mi)
#pragma unroll
      for (int ni = 0; ni < 4; ++ni)
        acc[mi][ni] = wmma_bf16(af[mi], bfr[ni], acc[mi][ni]);
    __syncthreads();
  }
#endif

#pragma unroll
  for (int mi = 0; mi < 4; ++mi)
#pragma unroll
    for (int ni = 0; ni < 4; ++ni)
#pragma unroll
      for (int r = 0; r < 8; ++r) {
        const int row = m0 + wm * 64 + mi * 16 + r + 8 * half;
        const int col = n0 + wn * 64 + ni * 16 + col0;
        if (col < N) C[(size_t)row * N + col] = (OutT)acc[mi][ni][r];
      }
}

// ---------------------------------------------------------------- qk-norm + rope

__global__ __launch_bounds__(256) void qkv_rope_kernel(
    const float* __restrict__ qf, const float* __restrict__ kf,
    const float* __restrict__ vf, const float* __restrict__ qn_w,
    const float* __restrict__ kn_w, bf16* __restrict__ Qr,
    bf16* __restrict__ Kr, bf16* __restrict__ Vr) {
  const int wid = threadIdx.x >> 5;
  const int lane = threadIdx.x & 31;
  const int idx = blockIdx.x * 8 + wid;          // over B*T*H
  const int hh = idx % H_;
  const int bt = idx / H_;                       // b*T + t
  const int tpos = bt % T_;
  const size_t inOff = (size_t)bt * D_ + (size_t)hh * HD_;
  const size_t outOff =
      ((size_t)(bt / T_) * H_ + hh) * (size_t)T_ * HD_ + (size_t)tpos * HD_;

  const float theta = (float)tpos * __expf(-(float)lane * (9.210340372f / 32.0f));
  float sn, cs;
  __sincosf(theta, &sn, &cs);

  {  // Q: head rmsnorm + rope
    float a = qf[inOff + lane], b = qf[inOff + lane + 32];
    float ss = a * a + b * b;
#pragma unroll
    for (int off = 16; off; off >>= 1) ss += __shfl_xor(ss, off);
    const float r = rsqrtf(ss * (1.0f / HD_) + 1e-6f);
    const float x1 = a * r * qn_w[lane];
    const float x2 = b * r * qn_w[lane + 32];
    Qr[outOff + lane] = (bf16)(x1 * cs - x2 * sn);
    Qr[outOff + lane + 32] = (bf16)(x2 * cs + x1 * sn);
  }
  {  // K
    float a = kf[inOff + lane], b = kf[inOff + lane + 32];
    float ss = a * a + b * b;
#pragma unroll
    for (int off = 16; off; off >>= 1) ss += __shfl_xor(ss, off);
    const float r = rsqrtf(ss * (1.0f / HD_) + 1e-6f);
    const float x1 = a * r * kn_w[lane];
    const float x2 = b * r * kn_w[lane + 32];
    Kr[outOff + lane] = (bf16)(x1 * cs - x2 * sn);
    Kr[outOff + lane + 32] = (bf16)(x2 * cs + x1 * sn);
  }
  Vr[outOff + lane] = (bf16)vf[inOff + lane];
  Vr[outOff + lane + 32] = (bf16)vf[inOff + lane + 32];
}

// ---------------------------------------------------------------- flash attention

__global__ __launch_bounds__(128) void flash_attn_kernel(
    const bf16* __restrict__ Qr, const bf16* __restrict__ Kr,
    const bf16* __restrict__ Vr, bf16* __restrict__ Ao) {
  __shared__ bf16 Ks[32 * 72];      // keys x hd (row-major, padded)
  __shared__ bf16 Vt[64 * 40];      // hd x keys (transposed, padded)
  __shared__ bf16 Pl[4][16 * 40];   // per-wave P tile 16x32

  const int bh = blockIdx.x;
  const int b = bh / H_, h = bh % H_;
  const int q0 = blockIdx.y * 64;
  const int t = threadIdx.x, lane = t & 31, wid = t >> 5;
  const int half = lane >> 4, col0 = lane & 15;
  const size_t base = (size_t)bh * T_ * HD_;

  const v16bf qfrag0 = load_frag(Qr + base + (size_t)(q0 + wid * 16) * HD_, HD_);
  const v16bf qfrag1 =
      load_frag(Qr + base + (size_t)(q0 + wid * 16) * HD_ + 32, HD_);

  v8f o[4];
#pragma unroll
  for (int n = 0; n < 4; ++n) o[n] = vzero8();
  float mrow[8], lrow[8];
#pragma unroll
  for (int r = 0; r < 8; ++r) { mrow[r] = -1e30f; lrow[r] = 0.0f; }

  const int ntiles = blockIdx.y * 2 + 2;  // keys up to q0+64 (causal)
  for (int kt = 0; kt < ntiles; ++kt) {
    const int k0 = kt * 32;
#pragma unroll
    for (int i = 0; i < 2; ++i) {
      const int c = t + i * 128;          // 256 chunks of 8 halves
      const int row = c >> 3, cid = c & 7;
      const v8bf kv =
          *(const v8bf*)(Kr + base + (size_t)(k0 + row) * HD_ + cid * 8);
      *(v8bf*)&Ks[row * 72 + cid * 8] = kv;
      const v8bf vv =
          *(const v8bf*)(Vr + base + (size_t)(k0 + row) * HD_ + cid * 8);
#pragma unroll
      for (int j = 0; j < 8; ++j) Vt[(cid * 8 + j) * 40 + row] = vv[j];
    }
    __syncthreads();

    v8f s0 = vzero8(), s1 = vzero8();
    s0 = wmma_bf16(qfrag0, load_frag(&Ks[0], 72), s0);
    s0 = wmma_bf16(qfrag1, load_frag(&Ks[32], 72), s0);
    s1 = wmma_bf16(qfrag0, load_frag(&Ks[16 * 72], 72), s1);
    s1 = wmma_bf16(qfrag1, load_frag(&Ks[16 * 72 + 32], 72), s1);

#pragma unroll
    for (int r = 0; r < 8; ++r) {
      const int rowq = q0 + wid * 16 + r + 8 * half;
      float a = s0[r] * 0.125f;           // 1/sqrt(64)
      float bb = s1[r] * 0.125f;
      if (k0 + col0 > rowq) a = -1e30f;
      if (k0 + 16 + col0 > rowq) bb = -1e30f;
      float mx = fmaxf(a, bb);
      mx = fmaxf(mx, __shfl_xor(mx, 1));
      mx = fmaxf(mx, __shfl_xor(mx, 2));
      mx = fmaxf(mx, __shfl_xor(mx, 4));
      mx = fmaxf(mx, __shfl_xor(mx, 8));
      const float mnew = fmaxf(mrow[r], mx);
      const float corr = __expf(mrow[r] - mnew);
      mrow[r] = mnew;
      const float p0 = __expf(a - mnew);
      const float p1 = __expf(bb - mnew);
      float ps = p0 + p1;
      ps += __shfl_xor(ps, 1);
      ps += __shfl_xor(ps, 2);
      ps += __shfl_xor(ps, 4);
      ps += __shfl_xor(ps, 8);
      lrow[r] = lrow[r] * corr + ps;
#pragma unroll
      for (int n = 0; n < 4; ++n) o[n][r] *= corr;
      Pl[wid][(r + 8 * half) * 40 + col0] = (bf16)p0;
      Pl[wid][(r + 8 * half) * 40 + col0 + 16] = (bf16)p1;
    }

    const v16bf pf = load_frag(&Pl[wid][0], 40);
#pragma unroll
    for (int n = 0; n < 4; ++n)
      o[n] = wmma_bf16(pf, load_frag(&Vt[n * 16 * 40], 40), o[n]);
    __syncthreads();
  }

#pragma unroll
  for (int r = 0; r < 8; ++r) {
    const int rowq = q0 + wid * 16 + r + 8 * half;
    const float inv = 1.0f / lrow[r];
#pragma unroll
    for (int n = 0; n < 4; ++n)
      Ao[((size_t)b * T_ + rowq) * D_ + h * HD_ + n * 16 + col0] =
          (bf16)(o[n][r] * inv);
  }
}

// ---------------------------------------------------------------- delta residual

__global__ __launch_bounds__(256) void delta_res_kernel(
    const float* __restrict__ x, const float* __restrict__ h,
    const float* __restrict__ gn_w, const float* __restrict__ g_w,
    const float* __restrict__ g_b, const float* __restrict__ wv_w,
    const float* __restrict__ mlp_w, float* __restrict__ x_out,
    bf16* __restrict__ xm_out, int write_xm) {
  const int row = blockIdx.x;
  const float* xr = x + (size_t)row * D_;
  const float* hr = h + (size_t)row * D_;
  float hh = 0, hx = 0, xx = 0, xg = 0, xw = 0;
  for (int i = threadIdx.x; i < D_; i += 256) {
    const float xi = xr[i], hi = hr[i];
    hh += hi * hi;
    hx += hi * xi;
    xx += xi * xi;
    xg += xi * gn_w[i] * g_w[i];
    xw += xi * wv_w[i];
  }
#pragma unroll
  for (int off = 16; off; off >>= 1) {
    hh += __shfl_xor(hh, off);
    hx += __shfl_xor(hx, off);
    xx += __shfl_xor(xx, off);
    xg += __shfl_xor(xg, off);
    xw += __shfl_xor(xw, off);
  }
  __shared__ float red[8][5];
  __shared__ float sc[3];
  const int lane = threadIdx.x & 31, wid = threadIdx.x >> 5;
  if (lane == 0) {
    red[wid][0] = hh; red[wid][1] = hx; red[wid][2] = xx;
    red[wid][3] = xg; red[wid][4] = xw;
  }
  __syncthreads();
  if (threadIdx.x == 0) {
    float HH = 0, HX = 0, XX = 0, XG = 0, XW = 0;
#pragma unroll
    for (int w = 0; w < 8; ++w) {
      HH += red[w][0]; HX += red[w][1]; XX += red[w][2];
      XG += red[w][3]; XW += red[w][4];
    }
    const float nrm = sqrtf(HH);
    const float kf = 1.0f / (fmaxf(nrm, 1e-6f) * 32.0f);  // * 1/sqrt(D)
    const float kTx = HX * kf;
    const float r = rsqrtf(XX * (1.0f / D_) + 1e-6f);
    const float beta = 2.0f * sigmoidf_(r * XG + g_b[0]);
    const float vv = sigmoidf_(XW);
    const float s = beta * (vv - kTx);
    sc[0] = kf;
    sc[1] = s;
    if (write_xm) {
      const float xk = HX * kf, kk = HH * kf * kf;
      const float xn2 = XX + 2.0f * s * xk + s * s * kk;
      sc[2] = rsqrtf(xn2 * (1.0f / D_) + 1e-6f);
    } else {
      sc[2] = 0.0f;
    }
  }
  __syncthreads();
  const float kf = sc[0], s = sc[1], r2 = sc[2];
  for (int i = threadIdx.x; i < D_; i += 256) {
    const float xn = xr[i] + s * kf * hr[i];
    x_out[(size_t)row * D_ + i] = xn;
    if (write_xm)
      xm_out[(size_t)row * D_ + i] = (bf16)(xn * r2 * mlp_w[i]);
  }
}

// ---------------------------------------------------------------- silu gate

__global__ void silu_gate_kernel(const bf16* __restrict__ a,
                                 const bf16* __restrict__ b,
                                 bf16* __restrict__ g, int n) {
  int i = blockIdx.x * blockDim.x + threadIdx.x;
  int st = gridDim.x * blockDim.x;
  for (; i < n; i += st) {
    const float av = (float)a[i], bv = (float)b[i];
    g[i] = (bf16)((av * sigmoidf_(av)) * bv);
  }
}

// ---------------------------------------------------------------- launch

extern "C" void kernel_launch(void* const* d_in, const int* in_sizes, int n_in,
                              void* d_out, int out_size, void* d_ws,
                              size_t ws_size, hipStream_t stream) {
  (void)in_sizes; (void)n_in; (void)out_size; (void)ws_size;
  const float* x           = (const float*)d_in[0];
  const float* wq          = (const float*)d_in[1];
  const float* wk          = (const float*)d_in[2];
  const float* wv          = (const float*)d_in[3];
  const float* wo          = (const float*)d_in[4];
  const float* qn_w        = (const float*)d_in[5];
  const float* kn_w        = (const float*)d_in[6];
  const float* attn_norm_w = (const float*)d_in[7];
  const float* g1_norm_w   = (const float*)d_in[8];
  const float* g1_w        = (const float*)d_in[9];
  const float* g1_b        = (const float*)d_in[10];
  const float* wv1         = (const float*)d_in[11];
  const float* mlp_norm_w  = (const float*)d_in[12];
  const float* w1          = (const float*)d_in[13];
  const float* w2          = (const float*)d_in[14];
  const float* w3          = (const float*)d_in[15];
  const float* g2_norm_w   = (const float*)d_in[16];
  const float* g2_w        = (const float*)d_in[17];
  const float* g2_b        = (const float*)d_in[18];
  const float* wv2         = (const float*)d_in[19];
  float* out = (float*)d_out;

  char* ws = (char*)d_ws;
  size_t off = 0;
  auto alloc = [&](size_t bytes) -> char* {
    char* p = ws + off;
    off += (bytes + 255) & ~(size_t)255;
    return p;
  };

  bf16* Wq = (bf16*)alloc((size_t)D_ * D_ * 2);
  bf16* Wk = (bf16*)alloc((size_t)D_ * D_ * 2);
  bf16* Wv = (bf16*)alloc((size_t)D_ * D_ * 2);
  bf16* Wo = (bf16*)alloc((size_t)D_ * D_ * 2);
  bf16* W1 = (bf16*)alloc((size_t)FH_ * D_ * 2);
  bf16* W3 = (bf16*)alloc((size_t)FH_ * D_ * 2);
  bf16* W2 = (bf16*)alloc((size_t)D_ * FH_ * 2);
  bf16* xcb = (bf16*)alloc((size_t)M_ * D_ * 2);
  float* qf = (float*)alloc((size_t)M_ * D_ * 4);
  float* kf = (float*)alloc((size_t)M_ * D_ * 4);
  float* vf = (float*)alloc((size_t)M_ * D_ * 4);
  bf16* Qr = (bf16*)alloc((size_t)M_ * D_ * 2);
  bf16* Kr = (bf16*)alloc((size_t)M_ * D_ * 2);
  bf16* Vr = (bf16*)alloc((size_t)M_ * D_ * 2);
  bf16* Ao = (bf16*)alloc((size_t)M_ * D_ * 2);
  float* h_attn = (float*)alloc((size_t)M_ * D_ * 4);
  float* x1 = (float*)alloc((size_t)M_ * D_ * 4);
  bf16* xm = (bf16*)alloc((size_t)M_ * D_ * 2);
  bf16* a_bf = (bf16*)alloc((size_t)M_ * FH_ * 2);
  bf16* b_bf = (bf16*)alloc((size_t)M_ * FH_ * 2);
  bf16* g_bf = (bf16*)alloc((size_t)M_ * FH_ * 2);
  float* h_mlp = (float*)alloc((size_t)M_ * D_ * 4);

  // 1) convert weights to bf16
  convert_f32_bf16<<<1024, 256, 0, stream>>>(wq, Wq, D_ * D_);
  convert_f32_bf16<<<1024, 256, 0, stream>>>(wk, Wk, D_ * D_);
  convert_f32_bf16<<<1024, 256, 0, stream>>>(wv, Wv, D_ * D_);
  convert_f32_bf16<<<1024, 256, 0, stream>>>(wo, Wo, D_ * D_);
  convert_f32_bf16<<<1024, 256, 0, stream>>>(w1, W1, FH_ * D_);
  convert_f32_bf16<<<1024, 256, 0, stream>>>(w3, W3, FH_ * D_);
  convert_f32_bf16<<<1024, 256, 0, stream>>>(w2, W2, D_ * FH_);

  // 2) attn rmsnorm
  rmsnorm_bf16_kernel<<<M_, 256, 0, stream>>>(x, attn_norm_w, xcb);

  // 3) QKV GEMMs
  const dim3 gD(D_ / 256, M_ / 128);
  gemm_bf16_kernel<float><<<gD, 256, 0, stream>>>(xcb, Wq, qf, D_, D_);
  gemm_bf16_kernel<float><<<gD, 256, 0, stream>>>(xcb, Wk, kf, D_, D_);
  gemm_bf16_kernel<float><<<gD, 256, 0, stream>>>(xcb, Wv, vf, D_, D_);

  // 4) head rmsnorm + rope + layout change
  qkv_rope_kernel<<<(B_ * T_ * H_) / 8, 256, 0, stream>>>(qf, kf, vf, qn_w,
                                                          kn_w, Qr, Kr, Vr);

  // 5) flash attention
  flash_attn_kernel<<<dim3(B_ * H_, T_ / 64), 128, 0, stream>>>(Qr, Kr, Vr, Ao);

  // 6) output projection
  gemm_bf16_kernel<float><<<gD, 256, 0, stream>>>(Ao, Wo, h_attn, D_, D_);

  // 7) delta residual 1 (also emits mlp rmsnorm in bf16)
  delta_res_kernel<<<M_, 256, 0, stream>>>(x, h_attn, g1_norm_w, g1_w, g1_b,
                                           wv1, mlp_norm_w, x1, xm, 1);

  // 8) MLP up projections (N = 2752, edge handled by TDM OOB / guards)
  const dim3 gF((FH_ + 255) / 256, M_ / 128);
  gemm_bf16_kernel<bf16><<<gF, 256, 0, stream>>>(xm, W1, a_bf, FH_, D_);
  gemm_bf16_kernel<bf16><<<gF, 256, 0, stream>>>(xm, W3, b_bf, FH_, D_);

  // 9) silu gate
  silu_gate_kernel<<<4096, 256, 0, stream>>>(a_bf, b_bf, g_bf, M_ * FH_);

  // 10) MLP down projection (K = 2752)
  gemm_bf16_kernel<float><<<gD, 256, 0, stream>>>(g_bf, W2, h_mlp, D_, FH_);

  // 11) delta residual 2 -> final output
  delta_res_kernel<<<M_, 256, 0, stream>>>(x1, h_mlp, g2_norm_w, g2_w, g2_b,
                                           wv2, nullptr, out, nullptr, 0);
}
